// AetheriusCoreBlock_32950989094985
// MI455X (gfx1250) — matrix operand order repaired and checked by reading, not compiled, not run on hardware
//
#include <hip/hip_runtime.h>
#include <hip/hip_bf16.h>
#include <math.h>

// ---------------------------------------------------------------------------
// AetheriusCoreBlock on MI455X (gfx1250): WMMA f16 GEMMs (f32 accumulate),
// double-buffered LDS staging, register-resident S6 scan, expert-bucketed
// top-2 MoE.
// ---------------------------------------------------------------------------

typedef __attribute__((ext_vector_type(16))) _Float16 v16h;
typedef __attribute__((ext_vector_type(8)))  float    v8f;
typedef __attribute__((ext_vector_type(4)))  _Float16 v4h;

#define C_B   2
#define C_T   512
#define C_D   1024
#define C_S   1024
#define C_N   16
#define C_E   16
#define C_TOK (C_B * C_T)     // 1024 tokens
#define C_PAIRS (C_TOK * 2)   // 2048 (token, k) pairs
#define PERM_CAP 2048

#define GBM 128
#define GBN 128
#define GBK 32
#define ASTR 36   // LDS row stride (halves) for A tile: 72B -> 18 banks, 8B aligned
#define WSTR 34   // LDS row stride (halves) for W tile (transposed): 68B -> 17 banks

union F16Frag { v16h v; unsigned u[8]; };

// ===========================================================================
// Generic WMMA GEMM:  C[M,N] = A[M,K] @ W[K,N] (+ bias[N])
//   mode 0: plain                (rows 0..M-1)
//   mode 1: MoE W1: gather A rows by token = perm>>1, scatter C rows by pairId
//   mode 2: MoE W2: gather A rows by pairId,          scatter C rows by pairId
// Per-expert W/bias selected by blockIdx.z when mode != 0.
// Double-buffered LDS; row maps resolved to pointers before the K loop.
// ===========================================================================
__global__ __launch_bounds__(256)
void aeth_gemm_wmma(const float* __restrict__ A,
                    const float* __restrict__ W,
                    const float* __restrict__ bias,
                    float* __restrict__ C,
                    int M, int N, int K, int mode,
                    const int* __restrict__ counts,
                    const int* __restrict__ perm)
{
    __shared__ __align__(16) _Float16 As[2][GBM * ASTR];
    __shared__ __align__(16) _Float16 Ws[2][GBN * WSTR];
    __shared__ int rmapA[GBM];
    __shared__ int rmapC[GBM];

    const int tid = threadIdx.x;
    const int e   = blockIdx.z;
    const int Me  = (mode == 0) ? M : counts[e];
    const int mbase = blockIdx.x * GBM;
    if (mbase >= Me) return;                     // uniform per block (EXEC stays full)

    const float* Wv = (mode == 0) ? W : (W + (size_t)e * K * N);
    const float* bv = bias ? ((mode == 0) ? bias : (bias + (size_t)e * N)) : nullptr;
    const int n0 = blockIdx.y * GBN;

    if (tid < GBM) {
        int r = mbase + tid;
        int a = -1, c = -1;
        if (r < Me) {
            if (mode == 0) { a = r; c = r; }
            else {
                int p = perm[e * PERM_CAP + r];
                c = p;
                a = (mode == 1) ? (p >> 1) : p;
            }
        }
        rmapA[tid] = a;
        rmapC[tid] = c;
    }
    __syncthreads();

    // ---- K-invariant loader slots: resolve row map to pointers ONCE ----
    const float* aptr[4]; int am[4], ak[4];
#pragma unroll
    for (int i = 0; i < 4; ++i) {
        int idx = tid + i * 256;          // 0..1023 float4 slots of the A tile
        int m   = idx >> 3;
        int k4  = (idx & 7) * 4;
        am[i] = m; ak[i] = k4;
        int ar = rmapA[m];
        aptr[i] = (ar >= 0) ? (A + (size_t)ar * K + k4) : nullptr;
    }
    const float* wptr[4]; int wk[4], wn[4];
#pragma unroll
    for (int i = 0; i < 4; ++i) {
        int idx = tid + i * 256;          // 0..1023 float4 slots of the W tile
        int k   = idx >> 5;
        int n4  = (idx & 31) * 4;
        wk[i] = k; wn[i] = n4;
        wptr[i] = Wv + (size_t)k * N + n0 + n4;
    }

    const int wave  = tid >> 5;
    const int lane  = tid & 31;
    const int waveM = wave >> 2;      // 0..1 -> 64-row slab
    const int waveN = wave & 3;       // 0..3 -> 32-col slab
    const int lm    = lane & 15;
    const int hsel  = lane >> 4;      // lane half select

    v8f acc[4][2];
#pragma unroll
    for (int mi = 0; mi < 4; ++mi)
#pragma unroll
        for (int ni = 0; ni < 2; ++ni) {
            v8f z = {0.f, 0.f, 0.f, 0.f, 0.f, 0.f, 0.f, 0.f};
            acc[mi][ni] = z;
        }

    float4 aval[4], wval[4];
    auto stage_global = [&](int k0) {
        const bool pf = (k0 + GBK < K);
#pragma unroll
        for (int i = 0; i < 4; ++i) {
            if (aptr[i]) {
                aval[i] = *(const float4*)(aptr[i] + k0);
                if (pf) __builtin_prefetch(aptr[i] + k0 + GBK, 0, 1);  // global_prefetch_b8
            } else {
                float4 z; z.x = z.y = z.z = z.w = 0.f; aval[i] = z;
            }
        }
#pragma unroll
        for (int i = 0; i < 4; ++i) {
            const float* s = wptr[i] + (size_t)k0 * N;
            wval[i] = *(const float4*)s;
            if (pf) __builtin_prefetch(s + (size_t)GBK * N, 0, 1);
        }
    };
    auto store_lds = [&](int buf) {
#pragma unroll
        for (int i = 0; i < 4; ++i) {
            v4h h4;
            h4[0] = (_Float16)aval[i].x; h4[1] = (_Float16)aval[i].y;
            h4[2] = (_Float16)aval[i].z; h4[3] = (_Float16)aval[i].w;
            *(v4h*)&As[buf][am[i] * ASTR + ak[i]] = h4;
        }
#pragma unroll
        for (int i = 0; i < 4; ++i) {      // transpose W tile into [n][k]
            Ws[buf][(wn[i] + 0) * WSTR + wk[i]] = (_Float16)wval[i].x;
            Ws[buf][(wn[i] + 1) * WSTR + wk[i]] = (_Float16)wval[i].y;
            Ws[buf][(wn[i] + 2) * WSTR + wk[i]] = (_Float16)wval[i].z;
            Ws[buf][(wn[i] + 3) * WSTR + wk[i]] = (_Float16)wval[i].w;
        }
    };

    // prologue: tile 0 into buffer 0
    stage_global(0);
    store_lds(0);
    __syncthreads();

    const int nk = K / GBK;
    for (int t = 0; t < nk; ++t) {
        const int cur = t & 1;
        const bool more = (t + 1 < nk);
        if (more) stage_global((t + 1) * GBK);   // global loads overlap WMMA below

        // ---- build fragments per CDNA5 16-bit WMMA lane layouts ----
        F16Frag afr[4];
#pragma unroll
        for (int mi = 0; mi < 4; ++mi) {
            const _Float16* ap = &As[cur][(waveM * 64 + mi * 16 + lm) * ASTR];
            const int kb = hsel * 8;   // lanes 0-15: K 0..7/16..23; 16-31: 8..15/24..31
#pragma unroll
            for (int v = 0; v < 8; ++v) {
                int k = ((v >= 4) ? 16 : 0) + kb + 2 * (v & 3);
                afr[mi].u[v] = *(const unsigned*)(ap + k);
            }
        }
        F16Frag bfr[2];
#pragma unroll
        for (int ni = 0; ni < 2; ++ni) {
            const _Float16* bp = &Ws[cur][(waveN * 32 + ni * 16 + lm) * WSTR];
            const int kb = hsel * 8;   // lanes 0-15: K 0..7/16..23; 16-31: 8..15/24..31
#pragma unroll
            for (int v = 0; v < 8; ++v)
                bfr[ni].u[v] = *(const unsigned*)(bp + ((v >= 4) ? 16 : 0) + kb + 2 * (v & 3));
        }

        // ---- 8 x v_wmma_f32_16x16x32_f16 per wave per K-tile ----
#pragma unroll
        for (int mi = 0; mi < 4; ++mi)
#pragma unroll
            for (int ni = 0; ni < 2; ++ni)
                acc[mi][ni] = __builtin_amdgcn_wmma_f32_16x16x32_f16(
                    false, afr[mi].v, false, bfr[ni].v,
                    (short)0, acc[mi][ni], false, false);

        if (more) store_lds((t + 1) & 1);        // write OTHER buffer: no collision
        __syncthreads();                         // one barrier per K-step
    }

    // ---- epilogue: C/D layout (VGPR r -> M = hsel*8 + r, N = lm) ----
    float bvals[2];
#pragma unroll
    for (int ni = 0; ni < 2; ++ni) {
        int col = n0 + waveN * 32 + ni * 16 + lm;
        bvals[ni] = bv ? bv[col] : 0.f;
    }
#pragma unroll
    for (int mi = 0; mi < 4; ++mi) {
#pragma unroll
        for (int ni = 0; ni < 2; ++ni) {
            int col = n0 + waveN * 32 + ni * 16 + lm;
#pragma unroll
            for (int r = 0; r < 8; ++r) {
                int rm = waveM * 64 + mi * 16 + hsel * 8 + r;
                int cr = rmapC[rm];
                if (cr >= 0)
                    C[(size_t)cr * N + col] = acc[mi][ni][r] + bvals[ni];
            }
        }
    }
}

// ===========================================================================
// Elementwise / reduction kernels
// ===========================================================================
__device__ __forceinline__ float sigf(float x) { return 1.f / (1.f + __expf(-x)); }

// p[BT,2D] -> xr = rotary(p[:, :D]) ; xg = sigmoid(p[:, D:])
__global__ void aeth_rotary(const float* __restrict__ p,
                            const float* __restrict__ snb,
                            const float* __restrict__ csb,
                            float* __restrict__ xr, float* __restrict__ xg)
{
    int i = blockIdx.x * 256 + threadIdx.x;          // over BT * D/2
    if (i >= C_TOK * (C_D / 2)) return;
    int hp = i & (C_D / 2 - 1);
    int bt = i >> 9;
    int t  = bt & (C_T - 1);
    const float* pr = p + (size_t)bt * (2 * C_D);
    float x1 = pr[2 * hp], x2 = pr[2 * hp + 1];
    float sn = snb[t * (C_D / 2) + hp], cs = csb[t * (C_D / 2) + hp];
    xr[(size_t)bt * C_D + 2 * hp]     = x1 * cs - x2 * sn;
    xr[(size_t)bt * C_D + 2 * hp + 1] = x1 * sn + x2 * cs;
    xg[(size_t)bt * C_D + 2 * hp]     = sigf(pr[C_D + 2 * hp]);
    xg[(size_t)bt * C_D + 2 * hp + 1] = sigf(pr[C_D + 2 * hp + 1]);
}

// depthwise conv (K=7, SAME over T) + silu * gate
__global__ void aeth_conv_gate(const float* __restrict__ xr,
                               const float* __restrict__ Wdw,
                               const float* __restrict__ bdw,
                               const float* __restrict__ xg,
                               float* __restrict__ t1)
{
    int i = blockIdx.x * 256 + threadIdx.x;           // over BT*D
    if (i >= C_TOK * C_D) return;
    int d  = i & (C_D - 1);
    int bt = i >> 10;
    int t  = bt & (C_T - 1);
    int b0 = bt - t;
    float acc = bdw[d];
#pragma unroll
    for (int j = 0; j < 7; ++j) {
        int tt = t + j - 3;
        if (tt >= 0 && tt < C_T)
            acc += xr[(size_t)(b0 + tt) * C_D + d] * Wdw[j * C_D + d];
    }
    float sl = acc * sigf(acc);
    t1[i] = sl * xg[i];
}

// RMS norm over D=1024 per token; optional residual add
__global__ void aeth_rms(const float* __restrict__ xin,
                         const float* __restrict__ g,
                         const float* __restrict__ res,
                         float* __restrict__ out)
{
    __shared__ float red[256];
    int token = blockIdx.x, t = threadIdx.x;
    const float* xp = xin + (size_t)token * C_D;
    float s = 0.f;
    for (int d = t; d < C_D; d += 256) { float v = xp[d]; s += v * v; }
    red[t] = s;
    __syncthreads();
    for (int o = 128; o > 0; o >>= 1) {
        if (t < o) red[t] += red[t + o];
        __syncthreads();
    }
    float r = 1.f / sqrtf(red[0] / (float)C_D + 1e-8f);
    for (int d = t; d < C_D; d += 256) {
        float v = g[d] * (xp[d] * r);
        out[(size_t)token * C_D + d] = res ? (res[(size_t)token * C_D + d] + v) : v;
    }
}

// S6 selective scan: one thread per (b, s) holds the 16-wide state in VGPRs.
__global__ void aeth_scan(const float* __restrict__ dpre,
                          const float* __restrict__ xi,
                          const float* __restrict__ Bm,
                          const float* __restrict__ Cm,
                          const float* __restrict__ Alog,
                          float* __restrict__ y)
{
    int gid = blockIdx.x * blockDim.x + threadIdx.x;    // B*S = 2048
    if (gid >= C_B * C_S) return;
    int b = gid >> 10, s = gid & (C_S - 1);
    float Arow[C_N], hst[C_N];
#pragma unroll
    for (int n = 0; n < C_N; ++n) {
        Arow[n] = -__expf(Alog[s * C_N + n]);
        hst[n]  = 0.f;
    }
    for (int t = 0; t < C_T; ++t) {
        size_t bt = (size_t)(b * C_T + t);
        float dp = dpre[bt * C_S + s];
        float d  = (dp > 20.f) ? dp : log1pf(__expf(dp));
        float u  = xi[bt * (2 * C_D) + s];
        const float4* Bp = (const float4*)(Bm + (bt * C_S + s) * C_N);
        const float4* Cp = (const float4*)(Cm + (bt * C_S + s) * C_N);
        float acc = 0.f;
#pragma unroll
        for (int q = 0; q < 4; ++q) {
            float4 Bv = Bp[q], Cv = Cp[q];
            float bb[4] = {Bv.x, Bv.y, Bv.z, Bv.w};
            float cc[4] = {Cv.x, Cv.y, Cv.z, Cv.w};
#pragma unroll
            for (int j = 0; j < 4; ++j) {
                int n = q * 4 + j;
                hst[n] = __expf(d * Arow[n]) * hst[n] + d * bb[j] * u;
                acc += hst[n] * cc[j];
            }
        }
        y[bt * C_S + s] = acc;
    }
}

// mix = sigmoid(g1)*s6_out + sigmoid(g2)*glcm_out
__global__ void aeth_mix(const float* __restrict__ g1, const float* __restrict__ g2,
                         const float* __restrict__ s6o, const float* __restrict__ gl,
                         float* __restrict__ mix)
{
    int i = blockIdx.x * 256 + threadIdx.x;
    if (i >= C_TOK * C_D) return;
    mix[i] = sigf(g1[i]) * s6o[i] + sigf(g2[i]) * gl[i];
}

// router: logits = h @ Wg (D x 16), top-2 + softmax
__global__ void aeth_gate_topk(const float* __restrict__ h,
                               const float* __restrict__ Wg,
                               float* __restrict__ wts, int* __restrict__ idx)
{
    __shared__ float part[16][17];
    int token = blockIdx.x, t = threadIdx.x;     // 256 threads
    int e = t & 15, dg = t >> 4;
    float s = 0.f;
    for (int d = dg; d < C_D; d += 16)
        s += h[(size_t)token * C_D + d] * Wg[d * C_E + e];
    part[dg][e] = s;
    __syncthreads();
    if (t < 16) {
        float tot = 0.f;
        for (int g = 0; g < 16; ++g) tot += part[g][t];   // fixed order: deterministic
        part[0][t] = tot;
    }
    __syncthreads();
    if (t == 0) {
        float v0 = -3.4e38f; int i0 = 0;
        for (int q = 0; q < C_E; ++q) if (part[0][q] > v0) { v0 = part[0][q]; i0 = q; }
        float v1 = -3.4e38f; int i1 = 0;
        for (int q = 0; q < C_E; ++q) if (q != i0 && part[0][q] > v1) { v1 = part[0][q]; i1 = q; }
        float e1 = __expf(v1 - v0);
        float z  = 1.f + e1;
        wts[token * 2 + 0] = 1.f / z;
        wts[token * 2 + 1] = e1 / z;
        idx[token * 2 + 0] = i0;
        idx[token * 2 + 1] = i1;
    }
}

__global__ void aeth_zero_i(int* __restrict__ p, int n)
{
    int i = blockIdx.x * blockDim.x + threadIdx.x;
    if (i < n) p[i] = 0;
}

// bucket (token,k) pairs by expert; order within bucket is irrelevant to values
__global__ void aeth_assign(const int* __restrict__ idx, int* __restrict__ counts,
                            int* __restrict__ perm)
{
    int p = blockIdx.x * blockDim.x + threadIdx.x;
    if (p >= C_PAIRS) return;
    int e = idx[p];
    int pos = atomicAdd(&counts[e], 1);
    perm[e * PERM_CAP + pos] = p;
}

__global__ void aeth_silu(float* __restrict__ x, int n)
{
    int i = blockIdx.x * 256 + threadIdx.x;
    if (i >= n) return;
    float v = x[i];
    x[i] = v * sigf(v);
}

// out = h + sum_k wts[k] * (e2[token*2+k] + moe_b2[idx[k]])
__global__ void aeth_combine(const float* __restrict__ h, const float* __restrict__ e2,
                             const float* __restrict__ b2, const float* __restrict__ wts,
                             const int* __restrict__ idx, float* __restrict__ out)
{
    int i = blockIdx.x * 256 + threadIdx.x;
    if (i >= C_TOK * C_D) return;
    int token = i >> 10, d = i & (C_D - 1);
    float w0 = wts[token * 2], w1 = wts[token * 2 + 1];
    int i0 = idx[token * 2], i1 = idx[token * 2 + 1];
    out[i] = h[i]
           + w0 * (e2[(size_t)(token * 2 + 0) * C_D + d] + b2[i0 * C_D + d])
           + w1 * (e2[(size_t)(token * 2 + 1) * C_D + d] + b2[i1 * C_D + d]);
}

// ===========================================================================
// Host orchestration
// ===========================================================================
extern "C" void kernel_launch(void* const* d_in, const int* in_sizes, int n_in,
                              void* d_out, int out_size, void* d_ws, size_t ws_size,
                              hipStream_t stream)
{
    const float* x       = (const float*)d_in[0];
    const float* snb     = (const float*)d_in[1];
    const float* csb     = (const float*)d_in[2];
    const float* glcm_Wp = (const float*)d_in[3];
    const float* glcm_bp = (const float*)d_in[4];
    const float* glcm_Wdw= (const float*)d_in[5];
    const float* glcm_bdw= (const float*)d_in[6];
    const float* glcm_g  = (const float*)d_in[7];
    const float* s6_Win  = (const float*)d_in[8];
    const float* s6_bin  = (const float*)d_in[9];
    const float* s6_Wd   = (const float*)d_in[10];
    const float* s6_bd   = (const float*)d_in[11];
    const float* s6_WB   = (const float*)d_in[12];
    const float* s6_bB   = (const float*)d_in[13];
    const float* s6_WC   = (const float*)d_in[14];
    const float* s6_bC   = (const float*)d_in[15];
    const float* s6_Alog = (const float*)d_in[16];
    const float* s6_Wout = (const float*)d_in[17];
    const float* s6_bout = (const float*)d_in[18];
    const float* s6_g    = (const float*)d_in[19];
    const float* gf_W1   = (const float*)d_in[20];
    const float* gf_b1   = (const float*)d_in[21];
    const float* gf_W2   = (const float*)d_in[22];
    const float* gf_b2   = (const float*)d_in[23];
    const float* gf_g    = (const float*)d_in[24];
    const float* moe_Wg  = (const float*)d_in[25];
    const float* moe_W1  = (const float*)d_in[26];
    const float* moe_b1  = (const float*)d_in[27];
    const float* moe_W2  = (const float*)d_in[28];
    const float* moe_b2  = (const float*)d_in[29];
    float* out = (float*)d_out;

    // ---- workspace carve-out (with buffer reuse) ----
    float* w = (float*)d_ws;
    size_t o = 0;
    auto alloc = [&](size_t n) { float* q = w + o; o += n; return q; };
    float* pbuf = alloc((size_t)C_TOK * 2 * C_D);     // proj p / later: h
    float* xr   = alloc((size_t)C_TOK * C_D);         // later: g1pre
    float* xg   = alloc((size_t)C_TOK * C_D);         // later: g2pre
    float* t1   = alloc((size_t)C_TOK * C_D);         // later: s6pre
    float* glcm = alloc((size_t)C_TOK * C_D);
    float* xi   = alloc((size_t)C_TOK * 2 * C_D);
    float* dpre = alloc((size_t)C_TOK * C_S);         // later: mix
    float* Bm   = alloc((size_t)C_TOK * C_S * C_N);   // later: e1 (2048x4096)
    float* Cm   = alloc((size_t)C_TOK * C_S * C_N);   // later: e2 (2048x1024)
    float* ybuf = alloc((size_t)C_TOK * C_S);
    float* s6o  = alloc((size_t)C_TOK * C_D);
    float* wts  = alloc(C_PAIRS);
    int*   idxb = (int*)alloc(C_PAIRS);
    int*   cnts = (int*)alloc(32);
    int*   perm = (int*)alloc((size_t)C_E * PERM_CAP);
    // aliases after lifetime ends
    float* hbuf  = pbuf;
    float* g1pre = xr;
    float* g2pre = xg;
    float* s6pre = t1;
    float* mixb  = dpre;
    float* e1    = Bm;
    float* e2    = Cm;

    dim3 blk(256);
    const int EW_TD = (C_TOK * C_D + 255) / 256;         // 4096 blocks

    // 1) p = x @ glcm_Wp + bp          [1024 x 2048]
    aeth_gemm_wmma<<<dim3(C_TOK / GBM, 2 * C_D / GBN, 1), blk, 0, stream>>>(
        x, glcm_Wp, glcm_bp, pbuf, C_TOK, 2 * C_D, C_D, 0, nullptr, nullptr);
    // 2) rotary + gate split
    aeth_rotary<<<(C_TOK * C_D / 2 + 255) / 256, blk, 0, stream>>>(pbuf, snb, csb, xr, xg);
    // 3) xi = x @ s6_Win + bin         [1024 x 2048]
    aeth_gemm_wmma<<<dim3(C_TOK / GBM, 2 * C_D / GBN, 1), blk, 0, stream>>>(
        x, s6_Win, s6_bin, xi, C_TOK, 2 * C_D, C_D, 0, nullptr, nullptr);
    // 4) depthwise conv + silu*gate
    aeth_conv_gate<<<EW_TD, blk, 0, stream>>>(xr, glcm_Wdw, glcm_bdw, xg, t1);
    // 5) glcm_out = rms(t1, glcm_g)
    aeth_rms<<<C_TOK, blk, 0, stream>>>(t1, glcm_g, nullptr, glcm);
    // 6) dpre = xi @ s6_Wd + bd        [1024 x 1024], K=2048
    aeth_gemm_wmma<<<dim3(C_TOK / GBM, C_S / GBN, 1), blk, 0, stream>>>(
        xi, s6_Wd, s6_bd, dpre, C_TOK, C_S, 2 * C_D, 0, nullptr, nullptr);
    // 7/8) Bm, Cm = xi @ s6_WB/WC      [1024 x 16384], K=2048
    aeth_gemm_wmma<<<dim3(C_TOK / GBM, C_S * C_N / GBN, 1), blk, 0, stream>>>(
        xi, s6_WB, s6_bB, Bm, C_TOK, C_S * C_N, 2 * C_D, 0, nullptr, nullptr);
    aeth_gemm_wmma<<<dim3(C_TOK / GBM, C_S * C_N / GBN, 1), blk, 0, stream>>>(
        xi, s6_WC, s6_bC, Cm, C_TOK, C_S * C_N, 2 * C_D, 0, nullptr, nullptr);
    // 9) selective scan -> y
    aeth_scan<<<(C_B * C_S + 255) / 256, blk, 0, stream>>>(dpre, xi, Bm, Cm, s6_Alog, ybuf);
    // 10) s6pre = y @ s6_Wout + bout
    aeth_gemm_wmma<<<dim3(C_TOK / GBM, C_D / GBN, 1), blk, 0, stream>>>(
        ybuf, s6_Wout, s6_bout, s6pre, C_TOK, C_D, C_S, 0, nullptr, nullptr);
    // 11) s6_out = rms(s6pre, s6_g)
    aeth_rms<<<C_TOK, blk, 0, stream>>>(s6pre, s6_g, nullptr, s6o);
    // 12/13) gate fusion pre-activations
    aeth_gemm_wmma<<<dim3(C_TOK / GBM, C_D / GBN, 1), blk, 0, stream>>>(
        x, gf_W1, gf_b1, g1pre, C_TOK, C_D, C_D, 0, nullptr, nullptr);
    aeth_gemm_wmma<<<dim3(C_TOK / GBM, C_D / GBN, 1), blk, 0, stream>>>(
        x, gf_W2, gf_b2, g2pre, C_TOK, C_D, C_D, 0, nullptr, nullptr);
    // 14) mix
    aeth_mix<<<EW_TD, blk, 0, stream>>>(g1pre, g2pre, s6o, glcm, mixb);
    // 15) h = x + rms(mix, gf_g)
    aeth_rms<<<C_TOK, blk, 0, stream>>>(mixb, gf_g, x, hbuf);
    // 16) router top-2
    aeth_gate_topk<<<C_TOK, blk, 0, stream>>>(hbuf, moe_Wg, wts, idxb);
    // 17/18) expert bucketing
    aeth_zero_i<<<1, 32, 0, stream>>>(cnts, 32);
    aeth_assign<<<(C_PAIRS + 255) / 256, blk, 0, stream>>>(idxb, cnts, perm);
    // 19) e1 = gather(h) @ W1[e] + b1[e]   (N=4096, K=1024)
    aeth_gemm_wmma<<<dim3(PERM_CAP / GBM, 4 * C_D / GBN, C_E), blk, 0, stream>>>(
        hbuf, moe_W1, moe_b1, e1, 0, 4 * C_D, C_D, 1, cnts, perm);
    // 20) silu(e1)
    aeth_silu<<<((C_PAIRS * 4 * C_D) + 255) / 256, blk, 0, stream>>>(e1, C_PAIRS * 4 * C_D);
    // 21) e2 = gather(e1) @ W2[e]          (N=1024, K=4096, bias folded into combine)
    aeth_gemm_wmma<<<dim3(PERM_CAP / GBM, C_D / GBN, C_E), blk, 0, stream>>>(
        e1, moe_W2, nullptr, e2, 0, C_D, 4 * C_D, 2, cnts, perm);
    // 22) out = h + weighted expert outputs (+ b2[e])
    aeth_combine<<<EW_TD, blk, 0, stream>>>(hbuf, e2, moe_b2, wts, idxb, out);

    (void)in_sizes; (void)n_in; (void)out_size; (void)ws_size;
}
